// bbgLSTM_32246614458882
// MI455X (gfx1250) — matrix operand order repaired and checked by reading, not compiled, not run on hardware
//
#include <hip/hip_runtime.h>

#define DD 1024   // hidden/input dim
#define BB 64     // batch
#define KK 512    // time steps
#define GD (4*DD) // gate dim

typedef __attribute__((ext_vector_type(16))) _Float16 v16h;
typedef __attribute__((ext_vector_type(8)))  _Float16 h8;
typedef __attribute__((ext_vector_type(8)))  float    v8f;
typedef __attribute__((ext_vector_type(4)))  float    f4;

#define WMMA_F16(A, B, C) \
  __builtin_amdgcn_wmma_f32_16x16x32_f16(false, (A), false, (B), (short)0, (C), false, false)

// ---------------------------------------------------------------------------
// Fragment loaders (wave32, 16x16x32 f16 WMMA).
//
// A (16x32, MxK), row-major source, this lane covers tile row (lane&15):
//   v16h elems 0..7  = K k+sel .. k+sel+7        (sel = 8 if lane>=16)
//   v16h elems 8..15 = K k+16+sel .. k+16+sel+7
// ---------------------------------------------------------------------------
__device__ __forceinline__ v16h a_frag_f16(const _Float16* __restrict__ rowp,
                                           int k, int hi) {
  const int sel = hi ? 8 : 0;
  h8 lo = *(const h8*)(rowp + k + sel);
  h8 up = *(const h8*)(rowp + k + 16 + sel);
  v16h r;
#pragma unroll
  for (int i = 0; i < 8; ++i) { r[i] = lo[i]; r[i + 8] = up[i]; }
  return r;
}

__device__ __forceinline__ v16h a_frag_f32(const float* __restrict__ rowp,
                                           int k, int hi) {
  const int sel = hi ? 8 : 0;
  f4 a0 = *(const f4*)(rowp + k + sel);
  f4 a1 = *(const f4*)(rowp + k + sel + 4);
  f4 b0 = *(const f4*)(rowp + k + 16 + sel);
  f4 b1 = *(const f4*)(rowp + k + 20 + sel);
  v16h r;
#pragma unroll
  for (int i = 0; i < 4; ++i) {
    r[i]      = (_Float16)a0[i];
    r[i + 4]  = (_Float16)a1[i];
    r[i + 8]  = (_Float16)b0[i];
    r[i + 12] = (_Float16)b1[i];
  }
  return r;
}

// B (32x16, KxN) with B[k][n] = W[n][k] (W row-major, row stride DD).
// Lane covers column n = lane&15; elems 0..7 = K k+8*hi .. +7, elems 8..15 = K k+16+8*hi .. +7.
__device__ __forceinline__ v16h b_frag_f16(const _Float16* __restrict__ rowp,
                                           int k, int hi) {
  const _Float16* p = rowp + k + (hi ? 8 : 0);
  h8 lo = *(const h8*)(p);
  h8 up = *(const h8*)(p + 16);
  v16h r;
#pragma unroll
  for (int i = 0; i < 8; ++i) { r[i] = lo[i]; r[i + 8] = up[i]; }
  return r;
}

__device__ __forceinline__ float sigmoid_f(float x) {
  return 1.0f / (1.0f + __expf(-x));
}

// One rotated fragment set: A tile + one B tile per gate.
struct Frags {
  v16h a, b0, b1, b2, b3;
};

// ---------------------------------------------------------------------------
// Software-pipelined K-loop: gates(+)= A(16xDD) x W^T for all 4 gate blocks.
// Next iteration's 5 fragments are loaded BEFORE the current 4 WMMAs are
// issued, so two fragment sets stay live and the loads of step k+1 overlap
// the WMMA chain of step k (waits become loadcnt>0 instead of 0).
// AF32: A sourced from fp32 x (converted on the fly); else f16 h.
// ---------------------------------------------------------------------------
template <bool AF32>
__device__ __forceinline__ void gemm_quad(
    const float*    __restrict__ xrow,   // AF32 path
    const _Float16* __restrict__ arow,   // f16 path
    const _Float16* __restrict__ p0, const _Float16* __restrict__ p1,
    const _Float16* __restrict__ p2, const _Float16* __restrict__ p3,
    int hi, v8f& ai, v8f& af, v8f& ag, v8f& ao)
{
  Frags cur;
  if constexpr (AF32) cur.a = a_frag_f32(xrow, 0, hi);
  else                cur.a = a_frag_f16(arow, 0, hi);
  cur.b0 = b_frag_f16(p0, 0, hi);
  cur.b1 = b_frag_f16(p1, 0, hi);
  cur.b2 = b_frag_f16(p2, 0, hi);
  cur.b3 = b_frag_f16(p3, 0, hi);

#pragma unroll 2
  for (int k = 32; k < DD; k += 32) {
    Frags nxt;
    if constexpr (AF32) nxt.a = a_frag_f32(xrow, k, hi);
    else                nxt.a = a_frag_f16(arow, k, hi);
    nxt.b0 = b_frag_f16(p0, k, hi);
    nxt.b1 = b_frag_f16(p1, k, hi);
    nxt.b2 = b_frag_f16(p2, k, hi);
    nxt.b3 = b_frag_f16(p3, k, hi);

    ai = WMMA_F16(cur.a, cur.b0, ai);
    af = WMMA_F16(cur.a, cur.b1, af);
    ag = WMMA_F16(cur.a, cur.b2, ag);
    ao = WMMA_F16(cur.a, cur.b3, ao);

    cur = nxt;
  }

  ai = WMMA_F16(cur.a, cur.b0, ai);
  af = WMMA_F16(cur.a, cur.b1, af);
  ag = WMMA_F16(cur.a, cur.b2, ag);
  ao = WMMA_F16(cur.a, cur.b3, ao);
}

// ---------------------------------------------------------------------------
// One LSTM step. PHASE1: gates = x@Wih^T + h@Whh^T + bias (K = 2*DD).
//                !PHASE1: gates = h@Wsum^T + bias          (K = DD).
// Each wave: one 16x16 (batch x d) tile, all 4 gates accumulated locally.
// Grid: 64 blocks x 128 threads = 256 waves = 4 row-tiles x 64 col-tiles.
// ---------------------------------------------------------------------------
template <bool PHASE1>
__global__ __launch_bounds__(128) void lstm_step(
    const float*    __restrict__ x,     // [BB,DD] fp32 (phase1 only)
    const _Float16* __restrict__ hin,   // [BB,DD] f16
    const float*    __restrict__ c0,    // [BB,DD] fp32
    const _Float16* __restrict__ W1,    // phase1: Wih16; phase2: Wsum16  [GD,DD]
    const _Float16* __restrict__ W2,    // phase1: Whh16; phase2: unused
    const float*    __restrict__ bias,  // [GD] = b_ih + b_hh
    _Float16*       __restrict__ hout,  // [BB,DD] f16 (next step input)
    float*          __restrict__ out32) // [BB,DD] fp32 or nullptr
{
  const int lane = threadIdx.x & 31;
  const int wave = blockIdx.x * (blockDim.x >> 5) + (threadIdx.x >> 5); // 0..255
  const int row0 = (wave & 3) * 16;   // batch tile base
  const int col0 = (wave >> 2) * 16;  // d tile base
  const int hi   = lane >> 4;
  const int ln   = lane & 15;

  const _Float16* hrow = hin + (size_t)(row0 + ln) * DD;

  // Per-gate W row pointers for this lane's output column.
  const _Float16* p0 = W1 + (size_t)(0 * DD + col0 + ln) * DD;
  const _Float16* p1 = W1 + (size_t)(1 * DD + col0 + ln) * DD;
  const _Float16* p2 = W1 + (size_t)(2 * DD + col0 + ln) * DD;
  const _Float16* p3 = W1 + (size_t)(3 * DD + col0 + ln) * DD;

  v8f acc_i = {}, acc_f = {}, acc_g = {}, acc_o = {};

  if constexpr (PHASE1) {
    const float* xrow = x + (size_t)(row0 + ln) * DD;
    gemm_quad<true>(xrow, nullptr, p0, p1, p2, p3, hi,
                    acc_i, acc_f, acc_g, acc_o);
    const _Float16* q0 = W2 + (size_t)(0 * DD + col0 + ln) * DD;
    const _Float16* q1 = W2 + (size_t)(1 * DD + col0 + ln) * DD;
    const _Float16* q2 = W2 + (size_t)(2 * DD + col0 + ln) * DD;
    const _Float16* q3 = W2 + (size_t)(3 * DD + col0 + ln) * DD;
    gemm_quad<false>(nullptr, hrow, q0, q1, q2, q3, hi,
                     acc_i, acc_f, acc_g, acc_o);
  } else {
    gemm_quad<false>(nullptr, hrow, p0, p1, p2, p3, hi,
                     acc_i, acc_f, acc_g, acc_o);
  }

  // Epilogue. C/D layout: lane column n = ln, VGPR r -> row r + 8*hi.
  const int d = col0 + ln;
  const float bi = bias[d];
  const float bf = bias[DD + d];
  const float bg = bias[2 * DD + d];
  const float bo = bias[3 * DD + d];

#pragma unroll
  for (int r = 0; r < 8; ++r) {
    const int b    = row0 + r + hi * 8;
    const float cv = c0[(size_t)b * DD + d];
    const float iv = sigmoid_f(acc_i[r] + bi);
    const float fv = sigmoid_f(acc_f[r] + bf);
    const float gv = tanhf(acc_g[r] + bg);
    const float ov = sigmoid_f(acc_o[r] + bo);
    const float cc = fv * cv + iv * gv;
    const float hv = ov * tanhf(cc);
    hout[(size_t)b * DD + d] = (_Float16)hv;
    if (out32) out32[(size_t)b * DD + d] = hv;
  }
}

// One-time weight prep: fp32 -> f16, W_sum = W_ih + W_hh, bias = b_ih + b_hh.
__global__ __launch_bounds__(256) void prep_weights_kernel(
    const float* __restrict__ Wih, const float* __restrict__ Whh,
    const float* __restrict__ bih, const float* __restrict__ bhh,
    _Float16* __restrict__ Wih16, _Float16* __restrict__ Whh16,
    _Float16* __restrict__ Wsum16, float* __restrict__ bias)
{
  const int i = blockIdx.x * blockDim.x + threadIdx.x; // grid == GD*DD exactly
  const float a = Wih[i];
  const float b = Whh[i];
  Wih16[i]  = (_Float16)a;
  Whh16[i]  = (_Float16)b;
  Wsum16[i] = (_Float16)(a + b);
  if (i < GD) bias[i] = bih[i] + bhh[i];
}

__global__ __launch_bounds__(256) void cvt_h0_kernel(
    const float* __restrict__ h0, _Float16* __restrict__ h16)
{
  const int i = blockIdx.x * blockDim.x + threadIdx.x; // grid == BB*DD exactly
  h16[i] = (_Float16)h0[i];
}

extern "C" void kernel_launch(void* const* d_in, const int* in_sizes, int n_in,
                              void* d_out, int out_size, void* d_ws, size_t ws_size,
                              hipStream_t stream) {
  (void)in_sizes; (void)n_in; (void)out_size; (void)ws_size;

  const float* batch = (const float*)d_in[0]; // [KK,BB,DD]
  const float* h0    = (const float*)d_in[1]; // [BB,DD]
  const float* c0    = (const float*)d_in[2]; // [BB,DD]
  const float* Wih   = (const float*)d_in[3]; // [GD,DD]
  const float* Whh   = (const float*)d_in[4]; // [GD,DD]
  const float* bih   = (const float*)d_in[5]; // [GD]
  const float* bhh   = (const float*)d_in[6]; // [GD]
  float* out = (float*)d_out;                 // [KK,BB,DD]

  // Workspace layout (~24.3 MB).
  char* ws = (char*)d_ws;
  size_t o = 0;
  _Float16* Wih16  = (_Float16*)(ws + o); o += (size_t)GD * DD * sizeof(_Float16);
  _Float16* Whh16  = (_Float16*)(ws + o); o += (size_t)GD * DD * sizeof(_Float16);
  _Float16* Wsum16 = (_Float16*)(ws + o); o += (size_t)GD * DD * sizeof(_Float16);
  float*    bias   = (float*)   (ws + o); o += (size_t)GD * sizeof(float);
  _Float16* hb[2];
  hb[0] = (_Float16*)(ws + o); o += (size_t)BB * DD * sizeof(_Float16);
  hb[1] = (_Float16*)(ws + o); o += (size_t)BB * DD * sizeof(_Float16);

  prep_weights_kernel<<<(GD * DD) / 256, 256, 0, stream>>>(
      Wih, Whh, bih, bhh, Wih16, Whh16, Wsum16, bias);
  cvt_h0_kernel<<<(BB * DD) / 256, 256, 0, stream>>>(h0, hb[0]);

  const dim3 grid(64), blk(128);
  int cur = 0;

  // Phase 1: 511 steps over batch[0..510]; last step's h is d_out[0].
  for (int t = 0; t < KK - 1; ++t) {
    float* o32 = (t == KK - 2) ? out : nullptr;
    lstm_step<true><<<grid, blk, 0, stream>>>(
        batch + (size_t)t * BB * DD, hb[cur], c0, Wih16, Whh16, bias,
        hb[cur ^ 1], o32);
    cur ^= 1;
  }

  // Phase 2: 511 self-feeding steps; outputs d_out[1..511].
  for (int t = 0; t < KK - 1; ++t) {
    lstm_step<false><<<grid, blk, 0, stream>>>(
        nullptr, hb[cur], c0, Wsum16, nullptr, bias,
        hb[cur ^ 1], out + (size_t)(t + 1) * BB * DD);
    cur ^= 1;
  }
}
